// StableTransformerEncoderLayerXL_26190710571289
// MI455X (gfx1250) — matrix operand order repaired and checked by reading, not compiled, not run on hardware
//
#include <hip/hip_runtime.h>
#include <stdint.h>

// ---------------------------------------------------------------------------
// Types for CDNA5 WMMA (wave32, V_WMMA_F32_16X16X32_BF16)
// ---------------------------------------------------------------------------
typedef __bf16 bf16_t;
typedef __attribute__((ext_vector_type(8)))  bf16_t bf16x8;
typedef __attribute__((ext_vector_type(16))) bf16_t bf16x16;
typedef __attribute__((ext_vector_type(8)))  float  v8f;

#define S_     1024
#define M_     1024
#define Bb     2
#define D_     1024
#define H_     16
#define DH_    64
#define DFF_   4096
#define HD_    1024          /* H_*DH_ */
#define SMQ    2048          /* S_+M_  */
#define ROWS   2048          /* S_*Bb  */
#define XROWS  4096          /* (S_+M_)*Bb */
#define SCALE_ 0.125f        /* 1/sqrt(64) */
#define SCR_LD 2064          /* padded score row stride (f32) */

static __device__ inline bf16_t f2bf(float x) {
  unsigned u = __float_as_uint(x);
  unsigned r = u + 0x7FFFu + ((u >> 16) & 1u);   // round-to-nearest-even
  unsigned short hs = (unsigned short)(r >> 16);
  bf16_t o; __builtin_memcpy(&o, &hs, 2); return o;
}
static __device__ inline v8f vzero() {
  v8f v;
  #pragma unroll
  for (int i = 0; i < 8; ++i) v[i] = 0.f;
  return v;
}
static __device__ inline v8f wmma_bf16(bf16x16 a, bf16x16 b, v8f c) {
  return __builtin_amdgcn_wmma_f32_16x16x32_bf16(false, a, false, b,
                                                 (short)0, c, false, false);
}
// A-fragment: lane holds row (lane&15); elems 0..15 -> K contiguous,
// K base = (lane>=16)?16:0  (matches B fragment order)
static __device__ inline bf16x16 load_a_frag(const bf16_t* p) {
  bf16x8 lo = *(const bf16x8*)(p);
  bf16x8 hi = *(const bf16x8*)(p + 8);
  return __builtin_shufflevector(lo, hi, 0,1,2,3,4,5,6,7,8,9,10,11,12,13,14,15);
}
// B-fragment: lane holds col (lane&15); elems 0..15 -> K contiguous,
// K base = (lane>=16)?16:0  (matches A fragment order)
static __device__ inline bf16x16 load_b_frag(const bf16_t* p) {
  bf16x8 lo = *(const bf16x8*)(p);
  bf16x8 hi = *(const bf16x8*)(p + 8);
  return __builtin_shufflevector(lo, hi, 0,1,2,3,4,5,6,7,8,9,10,11,12,13,14,15);
}
static __device__ inline float sigmf(float x) { return 1.f / (1.f + __expf(-x)); }

// ---------------------------------------------------------------------------
// Tensor Data Mover: 2D tile (tile_k x tile_rows, 2-byte elems) global->LDS,
// with LDS padding of 4 DWORDs (8 bf16) after every 16 DWORDs (32 bf16) so a
// 32-elem row lands on a 40-elem padded LDS row. D# layout per CDNA5 ISA §8.
// This toolchain's builtin takes 6 args (probe: u32x4, i32x8, i32x4, i32x4,
// i32x8, i32 cpol).
// ---------------------------------------------------------------------------
#if __has_builtin(__builtin_amdgcn_tensor_load_to_lds)
#define HAVE_TDM 1
typedef __attribute__((ext_vector_type(4))) unsigned int u32x4;
typedef __attribute__((ext_vector_type(8))) int          i32x8;
typedef __attribute__((ext_vector_type(4))) int          i32x4;

static __device__ __attribute__((always_inline)) inline void tdm_load_tile_2d(
    unsigned lds_byte_off, const void* gaddr, unsigned long long stride_elems)
{
  unsigned long long ga = (unsigned long long)(uintptr_t)gaddr;
  u32x4 g0;
  g0[0] = 1u;                                   // count=1 (valid user D#)
  g0[1] = lds_byte_off;                         // lds_addr
  g0[2] = (unsigned)(ga & 0xFFFFFFFFu);         // global_addr[31:0]
  g0[3] = (unsigned)((ga >> 32) & 0x01FFFFFFu)  // global_addr[56:32]
        | (2u << 30);                           // type=2 ("image")
  i32x8 g1;
  // [15:0]=wg_mask 0, [17:16]=data_size 1 (2B), [20]=pad_enable,
  // [24:22]=pad_interval 3 (16 DWORDs), [31:25]=pad_amount 3 (4 DWORDs)
  g1[0] = (int)((1u << 16) | (1u << 20) | (3u << 22) | (3u << 25));
  g1[1] = (int)0;                               // barrier addr 0; dim0[15:0]=0
  g1[2] = (int)0x00004000;                      // tensor_dim0 = 1<<30 (no OOB)
  g1[3] = (int)(0x00004000u | (32u << 16));     // tensor_dim1 = 1<<30; tile_dim0=32
  g1[4] = (int)128;                             // tile_dim1 = 128, tile_dim2 = 0
  g1[5] = (int)(unsigned)(stride_elems & 0xFFFFFFFFu);   // dim0_stride[31:0]
  g1[6] = (int)(unsigned)((stride_elems >> 32) & 0xFFFFu); // dim0_stride[47:32]
  g1[7] = 0;                                    // dim1_stride (unused, 2D)
  i32x4 gz4;
  gz4[0] = 0; gz4[1] = 0; gz4[2] = 0; gz4[3] = 0;  // groups 2/3: zero (2D tile)
  i32x8 gz8;
  #pragma unroll
  for (int i = 0; i < 8; ++i) gz8[i] = 0;
  __builtin_amdgcn_tensor_load_to_lds(g0, g1, gz4, gz4, gz8, 0);
}
#endif

// ---------------------------------------------------------------------------
// Generic bf16 WMMA GEMM: C[M,N] = A[M,K] @ B[K,N] (row-major, M%128==0,
// N%128==0, K%32==0). flags: 1=accumulate into Cin, 2=ReLU, 4=bf16 output.
// A tile (128x32) is staged via the Tensor Data Mover; B tile (32x128) is
// transposed manually into LDS so both fragment reads are contiguous b128.
// ---------------------------------------------------------------------------
__global__ __launch_bounds__(256) void gemm_bf16_kernel(
    const bf16_t* __restrict__ A, const bf16_t* __restrict__ Bm,
    const float* __restrict__ Cin, void* __restrict__ Cout,
    const float* __restrict__ bias, int Mn, int Nn, int Kn, int flags)
{
  __shared__ bf16_t Alds[128 * 40];   // 128 rows x 32 K (pad 8)
  __shared__ bf16_t Btld[128 * 40];   // 128 cols x 32 K (B transposed, pad 8)
  const int tid  = threadIdx.x;
  const int lane = tid & 31, wv = tid >> 5;
  const int wm = wv >> 1, wn = wv & 1;          // 4x2 wave grid
  const int lrow = lane & 15, hf = lane >> 4;
  const int rowBase = blockIdx.y * 128, colBase = blockIdx.x * 128;

  v8f acc[2][4];
  #pragma unroll
  for (int i = 0; i < 2; ++i)
    #pragma unroll
    for (int j = 0; j < 4; ++j) acc[i][j] = vzero();

  for (int k0 = 0; k0 < Kn; k0 += 32) {
#if defined(HAVE_TDM)
    if (wv == 0) {                               // one wave drives the DMA
      tdm_load_tile_2d((unsigned)(uintptr_t)(void*)&Alds[0],
                       &A[(size_t)rowBase * Kn + k0], (unsigned long long)Kn);
    }
#else
    for (int c = tid; c < 512; c += 256) {       // A tile: 128x32 (manual)
      int ar = c >> 2, kc = (c & 3) * 8;
      *(bf16x8*)(&Alds[ar * 40 + kc]) =
          *(const bf16x8*)(&A[(size_t)(rowBase + ar) * Kn + k0 + kc]);
    }
#endif
    if (k0 + 32 < Kn)
      __builtin_prefetch(&Bm[(size_t)(k0 + 32 + (tid & 31)) * Nn + colBase], 0, 1);
    for (int c = tid; c < 512; c += 256) {       // B tile: 32x128, transpose
      int kr = c >> 4, nc = (c & 15) * 8;
      bf16x8 vB = *(const bf16x8*)(&Bm[(size_t)(k0 + kr) * Nn + colBase + nc]);
      #pragma unroll
      for (int j = 0; j < 8; ++j) Btld[(nc + j) * 40 + kr] = vB[j];
    }
#if defined(HAVE_TDM)
    if (wv == 0) __builtin_amdgcn_s_wait_tensorcnt((short)0);
#endif
    __syncthreads();

    bf16x16 af[2], bfr[4];
    #pragma unroll
    for (int tm = 0; tm < 2; ++tm)
      af[tm] = load_a_frag(&Alds[(wm * 32 + tm * 16 + lrow) * 40 + hf * 16]);
    #pragma unroll
    for (int tn = 0; tn < 4; ++tn)
      bfr[tn] = load_b_frag(&Btld[(wn * 64 + tn * 16 + lrow) * 40 + hf * 16]);
    #pragma unroll
    for (int tm = 0; tm < 2; ++tm)
      #pragma unroll
      for (int tn = 0; tn < 4; ++tn)
        acc[tm][tn] = wmma_bf16(af[tm], bfr[tn], acc[tm][tn]);
    __syncthreads();
  }

  #pragma unroll
  for (int tm = 0; tm < 2; ++tm)
    #pragma unroll
    for (int tn = 0; tn < 4; ++tn)
      #pragma unroll
      for (int r = 0; r < 8; ++r) {
        int grow = rowBase + wm * 32 + tm * 16 + (hf ? r + 8 : r);
        int gcol = colBase + wn * 64 + tn * 16 + lrow;
        size_t idx = (size_t)grow * Nn + gcol;
        float val = acc[tm][tn][r];
        if (flags & 1) val += Cin[idx];
        if (bias)      val += bias[gcol];
        if (flags & 2) val = fmaxf(val, 0.f);
        if (flags & 4) ((bf16_t*)Cout)[idx] = f2bf(val);
        else           ((float*)Cout)[idx]  = val;
      }
}

// ---------------------------------------------------------------------------
// Fused rel-attention: per (b,h,16-query tile); scores row-block in LDS,
// rel_shift folded into a scatter-add, masked softmax, attn@V split-K.
// ---------------------------------------------------------------------------
__global__ __launch_bounds__(256) void attn_kernel(
    const bf16_t* __restrict__ qu, const bf16_t* __restrict__ qv,
    const bf16_t* __restrict__ kv, const bf16_t* __restrict__ hp,
    float* __restrict__ awv)
{
  extern __shared__ float smem[];
  float* scores = smem;                 // 16 * SCR_LD f32
  float* rowinv = smem + 16 * SCR_LD;   // 16 f32
  const int i0 = blockIdx.x * 16;
  const int h  = blockIdx.y;
  const int b  = blockIdx.z;
  const int tid = threadIdx.x, lane = tid & 31, wv = tid >> 5;
  const int lrow = lane & 15, hf = lane >> 4;

  // A fragments for this 16-query tile (q+u for content, q+v for position)
  const size_t qbase = ((size_t)((i0 + lrow) * Bb + b) * H_ + h) * DH_;
  bf16x16 aq[2], av[2];
  #pragma unroll
  for (int c = 0; c < 2; ++c) {
    aq[c] = load_a_frag(&qu[qbase + c * 32 + hf * 16]);
    av[c] = load_a_frag(&qv[qbase + c * 32 + hf * 16]);
  }

  // phase 1: content = (q+u) . k   (128 column tiles of 16, split over waves)
  for (int s = 0; s < 16; ++s) {
    int jt = wv * 16 + s;
    int jpos = jt * 16 + lrow;
    size_t kb = ((size_t)(jpos * Bb + b)) * 2048 + h * DH_;
    v8f acc = vzero();
    #pragma unroll
    for (int c = 0; c < 2; ++c)
      acc = wmma_bf16(aq[c], load_b_frag(&kv[kb + c * 32 + hf * 16]), acc);
    #pragma unroll
    for (int r = 0; r < 8; ++r) {
      int row = hf ? r + 8 : r;
      scores[row * SCR_LD + jt * 16 + lrow] = acc[r];
    }
  }
  __syncthreads();

  // phase 2: pos = (q+v) . hp, rel-shift fold: scores[i][r + i - (M-1)] += pos
  int rt0 = (M_ - 16 - i0) >> 4;
  for (int rt = rt0 + wv; rt < 128; rt += 8) {
    int rpos = rt * 16 + lrow;
    size_t hb = ((size_t)rpos * H_ + h) * DH_;
    v8f acc = vzero();
    #pragma unroll
    for (int c = 0; c < 2; ++c)
      acc = wmma_bf16(av[c], load_b_frag(&hp[hb + c * 32 + hf * 16]), acc);
    #pragma unroll
    for (int r = 0; r < 8; ++r) {
      int row = hf ? r + 8 : r;
      int i = i0 + row;
      int col = rt * 16 + lrow + i - (M_ - 1);
      if (col >= 0 && col < SMQ)
        scores[row * SCR_LD + col] += acc[r];  // bijective -> race-free
    }
  }
  __syncthreads();

  // phase 3: masked softmax (valid keys j <= i + M)
  {
    int row = tid >> 4, sub = tid & 15;
    int jmax = i0 + row + M_;
    float* sr = scores + row * SCR_LD;
    float mx = -3.0e38f;
    for (int c2 = sub; c2 < SMQ; c2 += 16)
      if (c2 <= jmax) mx = fmaxf(mx, sr[c2] * SCALE_);
    #pragma unroll
    for (int m = 8; m; m >>= 1) mx = fmaxf(mx, __shfl_xor(mx, m, 32));
    float sm = 0.f;
    for (int c2 = sub; c2 < SMQ; c2 += 16) {
      float e = (c2 <= jmax) ? __expf(sr[c2] * SCALE_ - mx) : 0.f;
      sr[c2] = e;
      sm += e;
    }
    #pragma unroll
    for (int m = 8; m; m >>= 1) sm += __shfl_xor(sm, m, 32);
    if (sub == 0) rowinv[row] = 1.f / sm;
  }
  __syncthreads();

  // phase 4: out = attn @ v  (K=2048 split over waves, 8 chunks of 32 each)
  v8f oacc[4];
  #pragma unroll
  for (int t = 0; t < 4; ++t) oacc[t] = vzero();
  float inv = rowinv[lrow];
  for (int t = 0; t < 8; ++t) {
    int kc = wv * 8 + t;
    const float* sp = scores + lrow * SCR_LD + kc * 32;
    bf16x16 a;
    #pragma unroll
    for (int j = 0; j < 8; ++j) {           // normalize + f32->bf16 on the fly
      a[j]     = f2bf(sp[hf * 16 + j] * inv);
      a[8 + j] = f2bf(sp[hf * 16 + 8 + j] * inv);
    }
    #pragma unroll
    for (int tn = 0; tn < 4; ++tn) {
      bf16x16 bb;
      #pragma unroll
      for (int j = 0; j < 16; ++j) {
        int jpos = kc * 32 + hf * 16 + j;
        bb[j] = kv[((size_t)(jpos * Bb + b)) * 2048 + 1024 + h * DH_ + tn * 16 + lrow];
      }
      oacc[tn] = wmma_bf16(a, bb, oacc[tn]);
    }
  }
  __syncthreads();
  float* part = scores;                     // reuse LDS for partials
  #pragma unroll
  for (int tn = 0; tn < 4; ++tn)
    #pragma unroll
    for (int r = 0; r < 8; ++r) {
      int row = hf ? r + 8 : r;
      part[wv * 1024 + row * 64 + tn * 16 + lrow] = oacc[tn][r];
    }
  __syncthreads();
  for (int o = tid; o < 1024; o += 256) {
    float ssum = 0.f;
    #pragma unroll
    for (int ww = 0; ww < 8; ++ww) ssum += part[ww * 1024 + o];
    int row = o >> 6, d = o & 63;
    awv[((size_t)((i0 + row) * Bb + b)) * HD_ + h * DH_ + d] = ssum;
  }
}

// ---------------------------------------------------------------------------
// Elementwise / LayerNorm helpers
// ---------------------------------------------------------------------------
__global__ __launch_bounds__(256) void cvt_bf16_kernel(
    const float* __restrict__ in, bf16_t* __restrict__ out, int n)
{
  int i = blockIdx.x * 256 + threadIdx.x;
  if (i < n) out[i] = f2bf(in[i]);
}

__global__ __launch_bounds__(256) void ln_bf16_kernel(
    const float* __restrict__ x, const float* __restrict__ g,
    const float* __restrict__ be, bf16_t* __restrict__ out)
{
  const int row = blockIdx.x;
  const float* xr = x + (size_t)row * D_;
  float s = 0.f, s2 = 0.f;
  for (int i = threadIdx.x; i < D_; i += 256) { float vv = xr[i]; s += vv; s2 += vv * vv; }
  __shared__ float r1[8], r2[8];
  #pragma unroll
  for (int m = 16; m; m >>= 1) { s += __shfl_xor(s, m, 32); s2 += __shfl_xor(s2, m, 32); }
  int lane = threadIdx.x & 31, wv = threadIdx.x >> 5;
  if (lane == 0) { r1[wv] = s; r2[wv] = s2; }
  __syncthreads();
  if (threadIdx.x == 0) {
    float a = 0.f, bq = 0.f;
    for (int i = 0; i < 8; ++i) { a += r1[i]; bq += r2[i]; }
    r1[0] = a; r2[0] = bq;
  }
  __syncthreads();
  float mu = r1[0] * (1.f / D_);
  float var = r2[0] * (1.f / D_) - mu * mu;
  float rs = rsqrtf(var + 1e-5f);
  bf16_t* orow = out + (size_t)row * D_;
  for (int i = threadIdx.x; i < D_; i += 256)
    orow[i] = f2bf((xr[i] - mu) * rs * g[i] + be[i]);
}

__global__ __launch_bounds__(256) void qbias_kernel(
    const float* __restrict__ q, const float* __restrict__ u,
    const float* __restrict__ v, bf16_t* __restrict__ qu,
    bf16_t* __restrict__ qv, int n)
{
  int i = blockIdx.x * 256 + threadIdx.x;
  if (i < n) {
    int hd = i & (HD_ - 1);
    float qq = q[i];
    qu[i] = f2bf(qq + u[hd]);
    qv[i] = f2bf(qq + v[hd]);
  }
}

__global__ __launch_bounds__(256) void gate_r_kernel(
    const float* __restrict__ rpre, const float* __restrict__ bWr,
    const float* __restrict__ bUr, const float* __restrict__ x,
    bf16_t* __restrict__ rx, int n)
{
  int i = blockIdx.x * 256 + threadIdx.x;
  if (i < n) {
    int c = i & (D_ - 1);
    float r = sigmf(rpre[i] + bWr[c] + bUr[c]);
    rx[i] = f2bf(r * x[i]);
  }
}

__global__ __launch_bounds__(256) void gate_final_kernel(
    const float* __restrict__ zpre, const float* __restrict__ gpre,
    const float* __restrict__ bWz, const float* __restrict__ bUz,
    const float* __restrict__ bWg, const float* __restrict__ bUg,
    const float* __restrict__ x, float* __restrict__ out, int n)
{
  int i = blockIdx.x * 256 + threadIdx.x;
  if (i < n) {
    int c = i & (D_ - 1);
    float z  = sigmf(zpre[i] + bWz[c] + bUz[c] - 0.1f);
    float hc = tanhf(gpre[i] + bWg[c] + bUg[c]);
    out[i] = (1.f - z) * x[i] + z * hc;
  }
}

// ---------------------------------------------------------------------------
// Host orchestration
// ---------------------------------------------------------------------------
extern "C" void kernel_launch(void* const* d_in, const int* in_sizes, int n_in,
                              void* d_out, int out_size, void* d_ws, size_t ws_size,
                              hipStream_t stream)
{
  (void)in_sizes; (void)n_in; (void)out_size; (void)ws_size;
  const float* encode = (const float*)d_in[0];
  const float* posf   = (const float*)d_in[1];
  const float* u      = (const float*)d_in[2];
  const float* vbias  = (const float*)d_in[3];
  const float* memory = (const float*)d_in[5];
  const float* Wkv = (const float*)d_in[6];
  const float* Wq  = (const float*)d_in[7];
  const float* Wps = (const float*)d_in[8];
  const float* Wo  = (const float*)d_in[9];
  const float* W1  = (const float*)d_in[10];
  const float* b1  = (const float*)d_in[11];
  const float* W2  = (const float*)d_in[12];
  const float* b2  = (const float*)d_in[13];
  const float* ln1g = (const float*)d_in[14];
  const float* ln1b = (const float*)d_in[15];
  const float* ln2g = (const float*)d_in[16];
  const float* ln2b = (const float*)d_in[17];
  // gate dicts flattened in insertion order: Wr,bWr,Ur,bUr,Wz,bWz,Uz,bUz,Wg,bWg,Ug,bUg
  auto gin = [&](int g, int k) -> const float* { return (const float*)d_in[18 + g * 12 + k]; };

  char* wsb = (char*)d_ws; size_t off = 0;
  auto alloc = [&](size_t bytes) -> void* {
    off = (off + 255) & ~(size_t)255; void* p = wsb + off; off += bytes; return p;
  };
  const size_t SBD = (size_t)ROWS * D_;   // 2M elems

  bf16_t* wWkv = (bf16_t*)alloc((size_t)D_ * 2048 * 2);
  bf16_t* wWq  = (bf16_t*)alloc((size_t)D_ * HD_ * 2);
  bf16_t* wWps = (bf16_t*)alloc((size_t)D_ * HD_ * 2);
  bf16_t* wWo  = (bf16_t*)alloc((size_t)HD_ * D_ * 2);
  bf16_t* wW1  = (bf16_t*)alloc((size_t)D_ * DFF_ * 2);
  bf16_t* wW2  = (bf16_t*)alloc((size_t)DFF_ * D_ * 2);
  bf16_t* gw[2][6];
  for (int g = 0; g < 2; ++g)
    for (int k = 0; k < 6; ++k) gw[g][k] = (bf16_t*)alloc((size_t)D_ * D_ * 2);
  bf16_t* pos_bf = (bf16_t*)alloc((size_t)SMQ * D_ * 2);
  bf16_t* xm_bf  = (bf16_t*)alloc((size_t)XROWS * D_ * 2);
  bf16_t* kv_bf  = (bf16_t*)alloc((size_t)XROWS * 2048 * 2);
  float*  qf     = (float*) alloc(SBD * 4);
  bf16_t* qu_bf  = (bf16_t*)alloc(SBD * 2);
  bf16_t* qv_bf  = (bf16_t*)alloc(SBD * 2);
  bf16_t* hp_bf  = (bf16_t*)alloc((size_t)SMQ * HD_ * 2);
  float*  awv_f  = (float*) alloc(SBD * 4);
  bf16_t* awv_bf = (bf16_t*)alloc(SBD * 2);
  float*  mha_f  = (float*) alloc(SBD * 4);
  bf16_t* x1_bf  = (bf16_t*)alloc(SBD * 2);
  bf16_t* y_bf   = (bf16_t*)alloc(SBD * 2);
  float*  rpre   = (float*) alloc(SBD * 4);
  float*  zpre   = (float*) alloc(SBD * 4);
  float*  gpre   = (float*) alloc(SBD * 4);
  bf16_t* rx_bf  = (bf16_t*)alloc(SBD * 2);
  float*  src_f  = (float*) alloc(SBD * 4);
  bf16_t* ln2_bf = (bf16_t*)alloc(SBD * 2);
  bf16_t* hid_bf = (bf16_t*)alloc((size_t)ROWS * DFF_ * 2);
  float*  ff_f   = (float*) alloc(SBD * 4);
  bf16_t* x2_bf  = (bf16_t*)alloc(SBD * 2);

  auto cvt = [&](const float* s, bf16_t* d, int n) {
    cvt_bf16_kernel<<<(n + 255) / 256, 256, 0, stream>>>(s, d, n);
  };
  auto gemm = [&](const bf16_t* A, const bf16_t* B, const float* Cin, void* Cout,
                  const float* bias, int Mn, int Nn, int Kn, int flags) {
    dim3 grid(Nn / 128, Mn / 128);
    gemm_bf16_kernel<<<grid, 256, 0, stream>>>(A, B, Cin, Cout, bias, Mn, Nn, Kn, flags);
  };

  // ---- weight conversions ----
  cvt(Wkv, wWkv, D_ * 2048);
  cvt(Wq,  wWq,  D_ * HD_);
  cvt(Wps, wWps, D_ * HD_);
  cvt(Wo,  wWo,  HD_ * D_);
  cvt(W1,  wW1,  D_ * DFF_);
  cvt(W2,  wW2,  DFF_ * D_);
  for (int g = 0; g < 2; ++g)
    for (int k = 0; k < 6; ++k) cvt(gin(g, 2 * k), gw[g][k], D_ * D_);

  // ---- activations: xm = [memory ; LN1(encode)] in bf16 ----
  cvt(posf,   pos_bf, SMQ * D_);
  cvt(memory, xm_bf,  M_ * Bb * D_);
  ln_bf16_kernel<<<ROWS, 256, 0, stream>>>(encode, ln1g, ln1b, xm_bf + (size_t)M_ * Bb * D_);
  cvt(encode, x1_bf, (int)SBD);

  // ---- projections ----
  gemm(xm_bf, wWkv, nullptr, kv_bf, nullptr, XROWS, 2048, D_, 4);                  // k,v (bf16)
  gemm(xm_bf + (size_t)M_ * Bb * D_, wWq, nullptr, qf, nullptr, ROWS, HD_, D_, 0); // q (f32)
  qbias_kernel<<<(int)((SBD + 255) / 256), 256, 0, stream>>>(qf, u, vbias, qu_bf, qv_bf, (int)SBD);
  gemm(pos_bf, wWps, nullptr, hp_bf, nullptr, SMQ, HD_, D_, 4);                    // hp (bf16)

  // ---- fused relative attention ----
  size_t smem = (size_t)(16 * SCR_LD + 16) * sizeof(float);
  attn_kernel<<<dim3(S_ / 16, H_, Bb), 256, smem, stream>>>(qu_bf, qv_bf, kv_bf, hp_bf, awv_f);

  // ---- output projection + gate 1 ----
  cvt(awv_f, awv_bf, (int)SBD);
  gemm(awv_bf, wWo, nullptr, mha_f, nullptr, ROWS, D_, HD_, 0);
  cvt(mha_f, y_bf, (int)SBD);

  auto run_gate = [&](int g, const float* xf, const bf16_t* xbf, const bf16_t* ybf, float* outp) {
    gemm(ybf, gw[g][0], nullptr, rpre, nullptr, ROWS, D_, D_, 0);   // y@Wr
    gemm(xbf, gw[g][1], rpre, rpre, nullptr, ROWS, D_, D_, 1);      // += x@Ur
    gemm(ybf, gw[g][2], nullptr, zpre, nullptr, ROWS, D_, D_, 0);   // y@Wz
    gemm(xbf, gw[g][3], zpre, zpre, nullptr, ROWS, D_, D_, 1);      // += x@Uz
    gemm(ybf, gw[g][4], nullptr, gpre, nullptr, ROWS, D_, D_, 0);   // y@Wg
    gate_r_kernel<<<(int)((SBD + 255) / 256), 256, 0, stream>>>(
        rpre, gin(g, 1), gin(g, 3), xf, rx_bf, (int)SBD);           // r, r*x
    gemm(rx_bf, gw[g][5], gpre, gpre, nullptr, ROWS, D_, D_, 1);    // += (r*x)@Ug
    gate_final_kernel<<<(int)((SBD + 255) / 256), 256, 0, stream>>>(
        zpre, gpre, gin(g, 5), gin(g, 7), gin(g, 9), gin(g, 11), xf, outp, (int)SBD);
  };
  run_gate(0, encode, x1_bf, y_bf, src_f);

  // ---- FFN + gate 2 ----
  ln_bf16_kernel<<<ROWS, 256, 0, stream>>>(src_f, ln2g, ln2b, ln2_bf);
  gemm(ln2_bf, wW1, nullptr, hid_bf, b1, ROWS, DFF_, D_, 2 | 4);    // relu, bf16
  gemm(hid_bf, wW2, nullptr, ff_f, b2, ROWS, D_, DFF_, 0);
  cvt(ff_f,  y_bf, (int)SBD);
  cvt(src_f, x2_bf, (int)SBD);
  run_gate(1, src_f, x2_bf, y_bf, (float*)d_out);
}
